// Ophiuchus_71554155151998
// MI455X (gfx1250) — matrix operand order repaired and checked by reading, not compiled, not run on hardware
//
#include <hip/hip_runtime.h>

typedef __attribute__((ext_vector_type(16))) __bf16 v16bf;
typedef __attribute__((ext_vector_type(8)))  float  v8f;

#define SC_IN   1600
#define SC_OUT  256
#define PPAD    24
#define EMB     32
#define VEC_OUT 64
#define FEAT    448            // 3*VEC_OUT + SC_OUT
#define MT      32             // residues per workgroup = 2 WMMA M tiles
#define MTILES  (MT/16)
#define KT      (SC_IN/32)     // 50 K-tiles of 32
#define NT      (SC_OUT/16)    // 16 N-tiles of 16

// ---------------------------------------------------------------------------
// Prep: repack W_sc (f32 [1600][256]) into bf16, laid out in exact WMMA-B
// fragment order so the main kernel's B loads are one contiguous 32B read
// per lane:  ws[((kt*NT+nt)*32+lane)*16+e]  <-  W_sc[K][n]
//   K = kt*32 + (lane<16 ? 0 : 16) + e ,  n = nt*16 + (lane&15)
// ---------------------------------------------------------------------------
__global__ void pack_wsc_bf16(const float* __restrict__ Wsc,
                              unsigned short* __restrict__ ws) {
  int idx = blockIdx.x * 256 + threadIdx.x;
  if (idx >= SC_IN * SC_OUT) return;
  int e    = idx & 15;
  int lane = (idx >> 4) & 31;
  int nt   = (idx >> 9) & 15;
  int kt   = idx >> 13;
  int K = kt * 32 + ((lane < 16) ? 0 : 16) + e;
  int n = nt * 16 + (lane & 15);
  __bf16 h = (__bf16)Wsc[K * SC_OUT + n];
  ws[idx] = *(const unsigned short*)&h;
}

// ---------------------------------------------------------------------------
// Fused kernel: one workgroup (256 threads = 8 waves) handles 32 residues
// (two 16-row WMMA A tiles). B fragments are reused across both A tiles in
// registers -> 4 v_wmma per 2 B loads per K-step.
// ---------------------------------------------------------------------------
__global__ __launch_bounds__(256)
void ophiuchus_fused(const float* __restrict__ base_coords,
                     const float* __restrict__ rel_coords,
                     const int*   __restrict__ rcode,
                     const int*   __restrict__ rseq,
                     const int*   __restrict__ acode,
                     const int*   __restrict__ atype,
                     const int*   __restrict__ ri,
                     const float* __restrict__ Wac,
                     const float* __restrict__ Wat,
                     const float* __restrict__ Wrc,
                     const float* __restrict__ Wri,
                     const float* __restrict__ Wvec,
                     const unsigned short* __restrict__ Bfrag,
                     float* __restrict__ out,
                     int R, int N) {
  __shared__ __bf16 sA[MTILES * KT * 32 * 16];   // 102400 B: A fragments
  __shared__ float  sWvec[PPAD * VEC_OUT];       // 6144 B
  __shared__ float  sRel[MT * PPAD * 3];         // 9216 B
  __shared__ int    sCode[MT * PPAD], sType[MT * PPAD];
  __shared__ int    sStart[MT], sCnt[MT];

  const int tid   = threadIdx.x;
  const int rbase = blockIdx.x * MT;

  // -- phase 0: per-residue atom ranges (ri is sorted, searchsorted) --------
  if (tid < MT) {
    int r = rbase + tid;
    int s = 0, c = 0;
    if (r < R) {
      int lo = 0, hi = N;
      while (lo < hi) { int mid = (lo + hi) >> 1; if (ri[mid] < r) lo = mid + 1; else hi = mid; }
      s = lo;
      lo = s; hi = N;
      while (lo < hi) { int mid = (lo + hi) >> 1; if (ri[mid] < r + 1) lo = mid + 1; else hi = mid; }
      c = lo - s;
      if (c > PPAD) c = PPAD;
    }
    sStart[tid] = s; sCnt[tid] = c;
  }
  // W_vec -> LDS (used by the l=1 path)
  for (int idx = tid; idx < PPAD * VEC_OUT; idx += 256) sWvec[idx] = Wvec[idx];
  __syncthreads();

  // -- phase 1: stage per-slot atom data ------------------------------------
  for (int idx = tid; idx < MT * PPAD; idx += 256) {
    int m = idx / PPAD, p = idx % PPAD;
    if (p < sCnt[m]) {
      int a = sStart[m] + p;
      sCode[idx] = acode[a];
      sType[idx] = atype[a];
      sRel[idx * 3 + 0] = rel_coords[a * 3 + 0];
      sRel[idx * 3 + 1] = rel_coords[a * 3 + 1];
      sRel[idx * 3 + 2] = rel_coords[a * 3 + 2];
    } else { sCode[idx] = -1; sType[idx] = -1; }
  }
  __syncthreads();

  // -- phase 2: scalar tile, pre-swizzled to WMMA-A layout ------------------
  for (int idx = tid; idx < MT * SC_IN; idx += 256) {
    int m = idx / SC_IN;
    int K = idx - m * SC_IN;
    int r = rbase + m;
    float val = 0.f;
    if (r < R) {
      if (K < PPAD * EMB) {                       // atom-code block
        int slot = K >> 5, c = K & 31;
        int code = sCode[m * PPAD + slot];
        if (code >= 0) val = Wac[code * EMB + c];
      } else if (K < 2 * PPAD * EMB) {            // atom-type block
        int K2 = K - PPAD * EMB;
        int slot = K2 >> 5, c = K2 & 31;
        int t = sType[m * PPAD + slot];
        if (t >= 0) val = Wat[t * EMB + c];
      } else if (K < 2 * PPAD * EMB + EMB) {      // residue-code block
        val = Wrc[rcode[r] * EMB + (K - 2 * PPAD * EMB)];
      } else {                                    // residue-index block
        val = Wri[rseq[r] * EMB + (K - 2 * PPAD * EMB - EMB)];
      }
    }
    // (m,K) -> A fragment (mtile, kt, lane, e), contiguous k = 16*hi + e (matches packed B)
    int kt = K >> 5, r32 = K & 31;
    int hi = r32 >> 4;
    int e  = r32 & 15;
    int lane  = (hi << 4) + (m & 15);
    int mtile = m >> 4;
    sA[((mtile * KT + kt) * 32 + lane) * 16 + e] = (__bf16)val;
  }

  // -- l=1 path: out_vec[r,mv,c] = (1/sqrt(P)) * sum_p rel[r,p,c]*Wvec[p,mv]
  for (int idx = tid; idx < MT * VEC_OUT * 3; idx += 256) {
    int m   = idx / (VEC_OUT * 3);
    int rem = idx - m * (VEC_OUT * 3);
    int mv  = rem / 3, c = rem % 3;
    int r = rbase + m;
    if (r < R) {
      int cnt = sCnt[m];
      float s = 0.f;
      for (int p = 0; p < cnt; ++p)
        s += sRel[(m * PPAD + p) * 3 + c] * sWvec[p * VEC_OUT + mv];
      out[(size_t)R * 3 + (size_t)r * FEAT + mv * 3 + c] = s * 0.20412414523193154f;
    }
  }

  // -- coords passthrough ---------------------------------------------------
  if (tid < MT * 3) {
    int r = rbase + tid / 3;
    if (r < R) out[r * 3 + tid % 3] = base_coords[r * 3 + tid % 3];
  }
  __syncthreads();

  // -- phase 3: GEMM via v_wmma_f32_16x16x32_bf16, 2 M-tiles x 2 N-tiles ----
  const int wave = tid >> 5, lane = tid & 31;
  const int nt0 = wave * 2, nt1 = wave * 2 + 1;
  v8f acc00 = {0.f,0.f,0.f,0.f,0.f,0.f,0.f,0.f};
  v8f acc01 = {0.f,0.f,0.f,0.f,0.f,0.f,0.f,0.f};
  v8f acc10 = {0.f,0.f,0.f,0.f,0.f,0.f,0.f,0.f};
  v8f acc11 = {0.f,0.f,0.f,0.f,0.f,0.f,0.f,0.f};
  const v16bf* A = (const v16bf*)sA;
  const v16bf* B = (const v16bf*)Bfrag;
  for (int kt = 0; kt < KT; ++kt) {
    v16bf a0 = A[(0 * KT + kt) * 32 + lane];            // 32B ds reads
    v16bf a1 = A[(1 * KT + kt) * 32 + lane];
    v16bf b0 = B[(kt * NT + nt0) * 32 + lane];          // 32B coalesced global
    v16bf b1 = B[(kt * NT + nt1) * 32 + lane];
    acc00 = __builtin_amdgcn_wmma_f32_16x16x32_bf16(false, a0, false, b0,
                                                    (short)0, acc00, false, false);
    acc01 = __builtin_amdgcn_wmma_f32_16x16x32_bf16(false, a0, false, b1,
                                                    (short)0, acc01, false, false);
    acc10 = __builtin_amdgcn_wmma_f32_16x16x32_bf16(false, a1, false, b0,
                                                    (short)0, acc10, false, false);
    acc11 = __builtin_amdgcn_wmma_f32_16x16x32_bf16(false, a1, false, b1,
                                                    (short)0, acc11, false, false);
  }

  // C layout: lanes 0-15 -> n=lane, M=v ; lanes 16-31 -> n=lane-16, M=v+8
  const int mbase = (lane >> 4) << 3;
  const int ncol  = lane & 15;
  const float scale = 0.025f;                           // 1/sqrt(1600)
  #pragma unroll
  for (int v = 0; v < 8; ++v) {
    int r0 = rbase + mbase + v;            // M-tile 0
    int r1 = r0 + 16;                      // M-tile 1
    if (r0 < R) {
      size_t o = (size_t)R * 3 + (size_t)r0 * FEAT + 3 * VEC_OUT;
      out[o + nt0 * 16 + ncol] = acc00[v] * scale;
      out[o + nt1 * 16 + ncol] = acc01[v] * scale;
    }
    if (r1 < R) {
      size_t o = (size_t)R * 3 + (size_t)r1 * FEAT + 3 * VEC_OUT;
      out[o + nt0 * 16 + ncol] = acc10[v] * scale;
      out[o + nt1 * 16 + ncol] = acc11[v] * scale;
    }
  }
}

// ---------------------------------------------------------------------------
extern "C" void kernel_launch(void* const* d_in, const int* in_sizes, int n_in,
                              void* d_out, int out_size, void* d_ws, size_t ws_size,
                              hipStream_t stream) {
  const float* base_coords = (const float*)d_in[0];
  const float* rel_coords  = (const float*)d_in[1];
  const int*   rcode       = (const int*)d_in[2];
  const int*   rseq        = (const int*)d_in[3];
  const int*   acode       = (const int*)d_in[4];
  const int*   atype       = (const int*)d_in[5];
  const int*   ri          = (const int*)d_in[6];
  const float* Wac         = (const float*)d_in[7];
  const float* Wat         = (const float*)d_in[8];
  const float* Wrc         = (const float*)d_in[9];
  const float* Wri         = (const float*)d_in[10];
  const float* Wvec        = (const float*)d_in[11];
  const float* Wsc         = (const float*)d_in[12];

  const int R = in_sizes[0] / 3;
  const int N = in_sizes[6];
  float* out = (float*)d_out;
  unsigned short* wsB = (unsigned short*)d_ws;   // SC_IN*SC_OUT bf16 = 800 KB

  // one-time-per-call repack of W_sc into bf16 WMMA-B fragment order
  int pack_blocks = (SC_IN * SC_OUT + 255) / 256;
  hipLaunchKernelGGL(pack_wsc_bf16, dim3(pack_blocks), dim3(256), 0, stream,
                     Wsc, wsB);

  int nblk = (R + MT - 1) / MT;                  // 1563 workgroups
  hipLaunchKernelGGL(ophiuchus_fused, dim3(nblk), dim3(256), 0, stream,
                     base_coords, rel_coords, rcode, rseq, acode, atype, ri,
                     Wac, Wat, Wrc, Wri, Wvec, wsB, out, R, N);
}
